// MarginMMD_Loss_9646496547400
// MI455X (gfx1250) — matrix operand order repaired and checked by reading, not compiled, not run on hardware
//
#include <hip/hip_runtime.h>

// MMD loss, MI455X (gfx1250, wave32, WMMA bf16 split-precision Gram).
// P=64 splits, each: A = [src_p; tgt_p] (128 x 2048 f32), G = A*A^T via
// bf16 hi/lo decomposition (G ~= hi*hiT + hi*loT + lo*hiT, fp32 accumulate).

#define PSPLITS 64
#define KSAMP   64
#define DDIM    2048
#define NROW    128
#define KC      32
#define NCHUNK  (DDIM / KC)

typedef __attribute__((ext_vector_type(16))) __bf16 v16bf;
typedef __attribute__((ext_vector_type(8)))  float  v8f;

struct B256 { uint4 a; uint4 b; };

__device__ __forceinline__ v16bf pack16(uint4 a, uint4 b) {
  B256 t; t.a = a; t.b = b;
  return __builtin_bit_cast(v16bf, t);
}

__device__ __forceinline__ unsigned short f2bf_rne(float x) {
  unsigned u = __float_as_uint(x);
  unsigned r = u + 0x7FFFu + ((u >> 16) & 1u);
  return (unsigned short)(r >> 16);
}
__device__ __forceinline__ float bf2f(unsigned short h) {
  return __uint_as_float(((unsigned)h) << 16);
}

__device__ __forceinline__ float wave_sum(float v) {
#pragma unroll
  for (int o = 16; o > 0; o >>= 1) v += __shfl_xor(v, o, 32);
  return v;
}
__device__ __forceinline__ float wave_max(float v) {
#pragma unroll
  for (int o = 16; o > 0; o >>= 1) v = fmaxf(v, __shfl_xor(v, o, 32));
  return v;
}

__global__ __launch_bounds__(256) void mmd_main(const float* __restrict__ src,
                                                const float* __restrict__ tgt,
                                                float* __restrict__ ws) {
  // LDS: chunk of A in bf16 hi/lo, row-major (A and B operands) and transposed (stored, never read).
  __shared__ __align__(16) unsigned short sAhi[NROW][KC];   // 8 KB
  __shared__ __align__(16) unsigned short sAlo[NROW][KC];   // 8 KB
  __shared__ __align__(16) unsigned short sThi[KC][NROW];   // 8 KB
  __shared__ __align__(16) unsigned short sTlo[KC][NROW];   // 8 KB
  __shared__ float sSq[NROW];
  __shared__ float sPart[256];
  __shared__ float sRed[16];   // [0..7] per-wave l2 sums, [8..15] per-wave l2 maxes
  __shared__ float sWq[16];    // [0..7] q(j<4) per wave, [8..15] q(j>=4) per wave

  const int p    = blockIdx.x;
  const int tid  = threadIdx.x;
  const int lane = tid & 31;
  const int wid  = tid >> 5;          // wave id = row-block i (0..7)

  // Loader mapping: 2 threads per row, 16 columns each per chunk.
  const int lr   = tid >> 1;          // 0..127
  const int half = tid & 1;
  const int col0 = half * 16;
  const float* rowbase = (lr < KSAMP)
      ? (src + (size_t)(p * KSAMP + lr) * DDIM)
      : (tgt + (size_t)(p * KSAMP + (lr - KSAMP)) * DDIM);

  v8f acc[8] = {};                    // 8 output tiles (16x16 f32) per wave
  float accsq = 0.0f;                 // fused row sum-of-squares partial

  // A-operand addressing (bf16 16x32 A layout):
  //   lanes 0-15: K {0..7} then {16..23}; lanes 16-31: K {8..15} then {24..31}
  const int ra = (wid << 4) + (lane & 15);
  const int kb = (lane & 16) ? 8 : 0;

  for (int chunk = 0; chunk < NCHUNK; ++chunk) {
    const int kc = chunk * KC;
    // ---- load 128x32 f32, decompose to bf16 hi/lo, stage both layouts ----
    const float* gp = rowbase + (size_t)kc + col0;
    if (chunk + 1 < NCHUNK) __builtin_prefetch(gp + KC, 0, 0);  // global_prefetch_b8
#pragma unroll
    for (int e = 0; e < 4; ++e) {
      float4 f = reinterpret_cast<const float4*>(gp)[e];
      float xs[4] = {f.x, f.y, f.z, f.w};
#pragma unroll
      for (int u = 0; u < 4; ++u) {
        const int c = col0 + e * 4 + u;
        const float x = xs[u];
        const unsigned short hi = f2bf_rne(x);
        const unsigned short lo = f2bf_rne(x - bf2f(hi));
        sAhi[lr][c] = hi;  sAlo[lr][c] = lo;
        sThi[c][lr] = hi;  sTlo[c][lr] = lo;
        accsq = fmaf(x, x, accsq);
      }
    }
    __syncthreads();

    // ---- WMMA: one 32-deep k-step, 8 column tiles, 3 products (hi/lo split) ----
    const uint4* aph = reinterpret_cast<const uint4*>(&sAhi[ra][kb]);
    const uint4* apl = reinterpret_cast<const uint4*>(&sAlo[ra][kb]);
    const v16bf a_hi = pack16(aph[0], aph[2]);   // segs at +0B and +32B within row
    const v16bf a_lo = pack16(apl[0], apl[2]);
#pragma unroll
    for (int j = 0; j < 8; ++j) {
      const uint4* bph = reinterpret_cast<const uint4*>(&sAhi[(j << 4) + (lane & 15)][kb]);
      const uint4* bpl = reinterpret_cast<const uint4*>(&sAlo[(j << 4) + (lane & 15)][kb]);
      const v16bf b_hi = pack16(bph[0], bph[2]);
      const v16bf b_lo = pack16(bpl[0], bpl[2]);
      acc[j] = __builtin_amdgcn_wmma_f32_16x16x32_bf16(false, a_hi, false, b_hi,
                                                       (short)0, acc[j], false, false);
      acc[j] = __builtin_amdgcn_wmma_f32_16x16x32_bf16(false, a_hi, false, b_lo,
                                                       (short)0, acc[j], false, false);
      acc[j] = __builtin_amdgcn_wmma_f32_16x16x32_bf16(false, a_lo, false, b_hi,
                                                       (short)0, acc[j], false, false);
    }
    __syncthreads();   // protect LDS before next chunk's stores
  }

  // ---- finish row norms: sq[r] = sum_d A[r][d]^2 ----
  sPart[tid] = accsq;
  __syncthreads();
  if (tid < NROW) sSq[tid] = sPart[2 * tid] + sPart[2 * tid + 1];
  __syncthreads();

  // ---- l2 = sq[r] + sq[c] - 2*G in registers; reduce sum and max ----
  // C/D layout: VGPR v, lanes 0-15 -> M=v, N=lane; lanes 16-31 -> M=v+8, N=lane-16
  const int rbase = (wid << 4) + ((lane & 16) ? 8 : 0);
  float lsum = 0.0f, lmax = -3.4e38f;
#pragma unroll
  for (int j = 0; j < 8; ++j) {
    const float sqc = sSq[(j << 4) + (lane & 15)];
#pragma unroll
    for (int v = 0; v < 8; ++v) {
      const float l2 = sSq[rbase + v] + sqc - 2.0f * acc[j][v];
      acc[j][v] = l2;
      lsum += l2;
      lmax = fmaxf(lmax, l2);
    }
  }
  lsum = wave_sum(lsum);
  lmax = wave_max(lmax);
  if (lane == 0) { sRed[wid] = lsum; sRed[8 + wid] = lmax; }
  __syncthreads();

  // bandwidth: bw = (sum(l2)/(n^2-n)) / KERNEL_MUL^(KERNEL_NUM/2), n=128
  float tot = 0.0f;
#pragma unroll
  for (int w = 0; w < 8; ++w) tot += sRed[w];
  const float bw = tot * (1.0f / 16256.0f) * 0.25f;
  float inv[5];
  {
    float mul = 1.0f;
#pragma unroll
    for (int m = 0; m < 5; ++m) { inv[m] = 1.0f / (bw * mul + 1e-9f); mul *= 2.0f; }
  }

  // ---- 5-kernel Gaussian sums into quadrant partials ----
  float q0 = 0.0f, q1 = 0.0f;   // j<4 (cols<64) / j>=4 (cols>=64)
#pragma unroll
  for (int j = 0; j < 8; ++j) {
#pragma unroll
    for (int v = 0; v < 8; ++v) {
      const float l2 = acc[j][v];
      float s = 0.0f;
#pragma unroll
      for (int m = 0; m < 5; ++m) s += __expf(-l2 * inv[m]);
      if (j < 4) q0 += s; else q1 += s;
    }
  }
  q0 = wave_sum(q0);
  q1 = wave_sum(q1);
  if (lane == 0) { sWq[wid] = q0; sWq[8 + wid] = q1; }
  __syncthreads();

  if (tid == 0) {
    // quad order: 0=XX (r<64,c<64), 1=YY, 2=XY (r<64,c>=64), 3=YX
    float quad[4] = {0.0f, 0.0f, 0.0f, 0.0f};
    for (int w = 0; w < 8; ++w) {
      const bool hi_i = (w >= 4);
      quad[hi_i ? 3 : 0] += sWq[w];       // j<4 half
      quad[hi_i ? 1 : 2] += sWq[8 + w];   // j>=4 half
    }
    float mx = sRed[8];
    for (int w = 1; w < 8; ++w) mx = fmaxf(mx, sRed[8 + w]);
    float* o = ws + p * 5;
    o[0] = quad[0]; o[1] = quad[1]; o[2] = quad[2]; o[3] = quad[3]; o[4] = mx;
  }
}

__global__ void mmd_finalize(const float* __restrict__ ws, float* __restrict__ out) {
  if (threadIdx.x == 0 && blockIdx.x == 0) {
    float sxx = 0.0f, syy = 0.0f, sxy = 0.0f, syx = 0.0f;
    for (int p = 0; p < PSPLITS; ++p) {
      sxx += ws[p * 5 + 0];
      syy += ws[p * 5 + 1];
      sxy += ws[p * 5 + 2];
      syx += ws[p * 5 + 3];
    }
    const float invq = 1.0f / 4096.0f;   // mean over 64x64 quadrant
    sxx *= invq; syy *= invq; sxy *= invq; syx *= invq;
    const float invP = 1.0f / (float)PSPLITS;
    out[0] = (sxx + syy - sxy - syx) * invP;
    out[1] = ws[(PSPLITS - 1) * 5 + 4];   // max(l2) of last split
    out[2] = sxx * invP;
    out[3] = syy * invP;
    out[4] = sxy * invP;
    out[5] = syx * invP;
  }
}

extern "C" void kernel_launch(void* const* d_in, const int* in_sizes, int n_in,
                              void* d_out, int out_size, void* d_ws, size_t ws_size,
                              hipStream_t stream) {
  (void)in_sizes; (void)n_in; (void)out_size; (void)ws_size;
  const float* src = (const float*)d_in[0];
  const float* tgt = (const float*)d_in[1];
  float* out = (float*)d_out;
  float* ws  = (float*)d_ws;           // 64 * 5 floats of per-split partials
  mmd_main<<<PSPLITS, 256, 0, stream>>>(src, tgt, ws);
  mmd_finalize<<<1, 32, 0, stream>>>(ws, out);
}
